// GNN_11416023073368
// MI455X (gfx1250) — compile-verified, matrix operand order checked by reading, not run on hardware
//
#include <hip/hip_runtime.h>
#include <hip/hip_bf16.h>

typedef __attribute__((ext_vector_type(16))) __bf16 v16bf;
typedef __attribute__((ext_vector_type(8)))  float  v8f;

#define H      128
#define LAYERS 4
#define EDIM   2
#define NDIM   28
#define TE     64   // edges per block iteration (4 row tiles of 16)
#define ROWT   4

__device__ __forceinline__ float gelu_exact(float v) {
    return 0.5f * v * (1.0f + erff(v * 0.70710678118654752f));
}

// fp32 -> bf16 bits, round-to-nearest-even
__device__ __forceinline__ unsigned short f2bf_bits(float f) {
    unsigned int u = __float_as_uint(f);
    unsigned int r = u + 0x7fffu + ((u >> 16) & 1u);
    return (unsigned short)(r >> 16);
}

// ---------------------------------------------------------------------------
// Pack cw[l][0:128][0:128] into WMMA B-operand layout:
// storage order [l][kchunk(4)][ntile(8)][lane(32)][i(16)]
//   k = kchunk*32 + (lane>>4)*16 + i      (K index, 0..127)
//   n = ntile*16 + (lane&15)              (N index, 0..127)
// ---------------------------------------------------------------------------
__global__ void pack_cw_kernel(const float* __restrict__ cw,
                               unsigned short* __restrict__ cwp) {
    int idx = blockIdx.x * blockDim.x + threadIdx.x;
    if (idx >= LAYERS * H * H) return;
    int i    = idx & 15;
    int lane = (idx >> 4) & 31;
    int nt   = (idx >> 9) & 7;
    int kc   = (idx >> 12) & 3;
    int l    = idx >> 14;
    int k = kc * 32 + ((lane >> 4) << 4) + i;
    int n = nt * 16 + (lane & 15);
    cwp[idx] = f2bf_bits(cw[((size_t)l * (H + EDIM) + k) * H + n]);
}

// ---------------------------------------------------------------------------
// Node embedding: h = gelu(LN(x @ emb_w + emb_b)); also bf16 mirror.
// ---------------------------------------------------------------------------
__global__ void embed_kernel(const float* __restrict__ x,
                             const float* __restrict__ w,
                             const float* __restrict__ b,
                             const float* __restrict__ g,
                             const float* __restrict__ beta,
                             float* __restrict__ h,
                             unsigned short* __restrict__ hb) {
    int nid = blockIdx.x;
    int t = threadIdx.x;                 // 0..127
    __shared__ float xr[NDIM];
    __shared__ float red[H];
    __shared__ float sh_mu, sh_var;
    if (t < NDIM) xr[t] = x[(size_t)nid * NDIM + t];
    __syncthreads();
    float acc = b[t];
    #pragma unroll
    for (int k = 0; k < NDIM; ++k) acc += xr[k] * w[k * H + t];
    red[t] = acc; __syncthreads();
    for (int s = 64; s > 0; s >>= 1) { if (t < s) red[t] += red[t + s]; __syncthreads(); }
    if (t == 0) sh_mu = red[0] * (1.0f / H);
    __syncthreads();
    float d = acc - sh_mu;
    red[t] = d * d; __syncthreads();
    for (int s = 64; s > 0; s >>= 1) { if (t < s) red[t] += red[t + s]; __syncthreads(); }
    if (t == 0) sh_var = red[0] * (1.0f / H);
    __syncthreads();
    float y  = d * rsqrtf(sh_var + 1e-5f) * g[t] + beta[t];
    float hv = gelu_exact(y);
    h[(size_t)nid * H + t]  = hv;
    hb[(size_t)nid * H + t] = f2bf_bits(hv);
}

// ---------------------------------------------------------------------------
// Edge message GEMM via WMMA (bf16 -> fp32), fused scatter-add.
// Persistent 256-thread blocks (8 wave32). Per iteration: 64 edges.
//   * async global->LDS gather of 64 rows of h_bf16 (double-buffered, 2x16KB)
//   * wave w owns column tile w; B fragments register-resident for the
//     whole kernel; 4 row tiles x 4 K-chunks = 16 wmma_f32_16x16x32_bf16
//   * epilogue: + bias + ea rank-2 term, exact GELU, atomic scatter
// ---------------------------------------------------------------------------
__global__ void msg_kernel(const unsigned short* __restrict__ hb,
                           const int* __restrict__ ei,     // [2,E]
                           const float* __restrict__ ea,   // [E,2]
                           const float* __restrict__ cw,   // [L,130,128] fp32
                           const float* __restrict__ cb,   // [L,128]
                           const unsigned short* __restrict__ cwp, // packed B
                           float* __restrict__ aout,       // [N,128]
                           long long E, int etiles, int layer) {
    __shared__ __align__(32) unsigned short Alds[2][TE * H];   // 2 x 16 KB

    const int tid   = threadIdx.x;
    const int lane  = tid & 31;
    const int ntile = tid >> 5;        // 0..7: column tile owned by this wave

    // B fragments: loaded once, live in registers for the whole kernel
    const unsigned short* bbase =
        cwp + ((size_t)(((layer * 4) * 8 + ntile) * 32) + lane) * 16;
    v16bf bfrag[4];
    #pragma unroll
    for (int kc = 0; kc < 4; ++kc)
        bfrag[kc] = *(const v16bf*)(bbase + (size_t)kc * 4096);

    const int col     = lane & 15;
    const int arow    = lane & 15;
    const int koff    = (lane >> 4) * 16;
    const int rowbase = (lane >> 4) * 8;
    const int n       = ntile * 16 + col;
    const float bias  = cb[layer * H + n];
    const float w0    = cw[((size_t)layer * (H + EDIM) + H) * H + n];
    const float w1    = cw[((size_t)layer * (H + EDIM) + H + 1) * H + n];

    // async gather mapping: thread -> (row = tid/4, 64B chunk = tid%4)
    const int prow = tid >> 2;          // 0..63
    const int pseg = (tid & 3) * 64;    // byte offset within the 256B row

    const long long t0     = blockIdx.x;
    const long long stride = gridDim.x;

    // issue async gather of edge-tile t into LDS buffer b
    auto prefetch = [&](long long t, int b) {
        long long e = t * TE + prow;
        int node = (e < E) ? ei[e] : 0;
        const char* gsrc =
            (const char*)(hb + (size_t)node * H) + pseg;
        unsigned ldst =
            (unsigned)(unsigned long long)(uintptr_t)(&Alds[b][prow * H]) + pseg;
        asm volatile(
            "global_load_async_to_lds_b128 %0, %1, off\n\t"
            "global_load_async_to_lds_b128 %0, %1, off offset:16\n\t"
            "global_load_async_to_lds_b128 %0, %1, off offset:32\n\t"
            "global_load_async_to_lds_b128 %0, %1, off offset:48"
            :: "v"(ldst), "v"(gsrc) : "memory");
    };

    int buf = 0;
    if (t0 < etiles) prefetch(t0, 0);

    for (long long t = t0; t < etiles; t += stride) {
        asm volatile("s_wait_asynccnt 0x0" ::: "memory");
        __syncthreads();

        long long tn = t + stride;                 // block-uniform predicate
        if (tn < etiles) prefetch(tn, buf ^ 1);

        const long long e0 = t * TE;
        #pragma unroll
        for (int rt = 0; rt < ROWT; ++rt) {
            v8f acc = {};
            #pragma unroll
            for (int kc = 0; kc < 4; ++kc) {
                v16bf av = *(const v16bf*)(
                    &Alds[buf][(rt * 16 + arow) * H + kc * 32 + koff]);
                acc = __builtin_amdgcn_wmma_f32_16x16x32_bf16(
                          false, av, false, bfrag[kc], (short)0, acc,
                          false, false);
            }
            #pragma unroll
            for (int j = 0; j < 8; ++j) {
                long long e = e0 + rt * 16 + rowbase + j;
                if (e < E) {
                    float v  = acc[j] + bias
                             + ea[e * 2 + 0] * w0 + ea[e * 2 + 1] * w1;
                    float mv = gelu_exact(v);
                    int dst  = ei[(size_t)E + e];
                    unsafeAtomicAdd(&aout[(size_t)dst * H + n], mv);
                }
            }
        }
        buf ^= 1;
        __syncthreads();   // all reads of old buffer done before its refill
    }
}

// ---------------------------------------------------------------------------
// h = LN(h + a); refresh bf16 mirror; fused per-graph pooling atomics.
// ---------------------------------------------------------------------------
__global__ void ln_res_kernel(float* __restrict__ h,
                              unsigned short* __restrict__ hb,
                              const float* __restrict__ a,
                              const float* __restrict__ g,
                              const float* __restrict__ b,
                              const int* __restrict__ batch,
                              float* __restrict__ pooled,
                              int layer, int B) {
    int nid = blockIdx.x;
    int t = threadIdx.x;                 // 0..127
    __shared__ float red[H];
    __shared__ float sh_mu, sh_var;
    float v = h[(size_t)nid * H + t] + a[(size_t)nid * H + t];
    red[t] = v; __syncthreads();
    for (int s = 64; s > 0; s >>= 1) { if (t < s) red[t] += red[t + s]; __syncthreads(); }
    if (t == 0) sh_mu = red[0] * (1.0f / H);
    __syncthreads();
    float d = v - sh_mu;
    red[t] = d * d; __syncthreads();
    for (int s = 64; s > 0; s >>= 1) { if (t < s) red[t] += red[t + s]; __syncthreads(); }
    if (t == 0) sh_var = red[0] * (1.0f / H);
    __syncthreads();
    float y = d * rsqrtf(sh_var + 1e-5f) * g[layer * H + t] + b[layer * H + t];
    h[(size_t)nid * H + t]  = y;
    hb[(size_t)nid * H + t] = f2bf_bits(y);
    int bi = batch[nid];
    unsafeAtomicAdd(&pooled[((size_t)layer * B + bi) * H + t], y);
}

__global__ void count_kernel(const int* __restrict__ batch,
                             float* __restrict__ counts, int N) {
    int i = blockIdx.x * blockDim.x + threadIdx.x;
    if (i < N) unsafeAtomicAdd(&counts[batch[i]], 1.0f);
}

// ---------------------------------------------------------------------------
// Head MLP (tiny: ~33 MFLOP total, plain VALU)
// ---------------------------------------------------------------------------
__global__ void head1_kernel(const float* __restrict__ pooled,
                             const float* __restrict__ counts,
                             const float* __restrict__ hw1,
                             const float* __restrict__ hb1,
                             float* __restrict__ z1, int B) {
    int bb = blockIdx.x; int t = threadIdx.x;   // 128 threads
    float inv = 1.0f / fmaxf(counts[bb], 1.0f);
    float acc = hb1[t];
    for (int l = 0; l < LAYERS; ++l) {
        const float* pr = pooled + ((size_t)l * B + bb) * H;
        for (int c = 0; c < H; ++c)
            acc += pr[c] * inv * hw1[((size_t)l * H + c) * H + t];
    }
    z1[(size_t)bb * H + t] = gelu_exact(acc);
}

__global__ void head2_kernel(const float* __restrict__ z1,
                             const float* __restrict__ hw2,
                             const float* __restrict__ hb2,
                             float* __restrict__ z2) {
    int bb = blockIdx.x; int t = threadIdx.x;   // 64 threads
    float acc = hb2[t];
    for (int c = 0; c < H; ++c)
        acc += z1[(size_t)bb * H + c] * hw2[c * (H / 2) + t];
    z2[(size_t)bb * (H / 2) + t] = gelu_exact(acc);
}

__global__ void head3_kernel(const float* __restrict__ z2,
                             const float* __restrict__ hw3,
                             const float* __restrict__ hb3,
                             float* __restrict__ out) {
    int bb = blockIdx.x; int t = threadIdx.x;   // 64 threads
    __shared__ float red[H / 2];
    red[t] = z2[(size_t)bb * (H / 2) + t] * hw3[t];
    __syncthreads();
    for (int s = 32; s > 0; s >>= 1) { if (t < s) red[t] += red[t + s]; __syncthreads(); }
    if (t == 0) out[bb] = red[0] + hb3[0];
}

// ---------------------------------------------------------------------------
extern "C" void kernel_launch(void* const* d_in, const int* in_sizes, int n_in,
                              void* d_out, int out_size, void* d_ws, size_t ws_size,
                              hipStream_t stream) {
    const float* x       = (const float*)d_in[0];
    const int*   ei      = (const int*)  d_in[1];
    const float* ea      = (const float*)d_in[2];
    const int*   batch   = (const int*)  d_in[3];
    const float* emb_w   = (const float*)d_in[4];
    const float* emb_b   = (const float*)d_in[5];
    const float* emb_lng = (const float*)d_in[6];
    const float* emb_lnb = (const float*)d_in[7];
    const float* cw      = (const float*)d_in[8];
    const float* cb      = (const float*)d_in[9];
    const float* ng      = (const float*)d_in[10];
    const float* nb      = (const float*)d_in[11];
    const float* hw1     = (const float*)d_in[12];
    const float* hb1     = (const float*)d_in[13];
    const float* hw2     = (const float*)d_in[14];
    const float* hb2     = (const float*)d_in[15];
    const float* hw3     = (const float*)d_in[16];
    const float* hb3     = (const float*)d_in[17];
    float* out = (float*)d_out;

    const int N = in_sizes[3];
    const long long E = in_sizes[2] / EDIM;
    const int B = out_size;

    char* p = (char*)d_ws;
    auto alloc = [&](size_t bytes) {
        char* r = p;
        p += (bytes + 255) & ~(size_t)255;
        return r;
    };
    float*          h      = (float*)         alloc((size_t)N * H * 4);
    unsigned short* hbm    = (unsigned short*)alloc((size_t)N * H * 2);
    float*          abuf   = (float*)         alloc((size_t)N * H * 4);
    float*          pooled = (float*)         alloc((size_t)LAYERS * B * H * 4);
    float*          counts = (float*)         alloc((size_t)B * 4);
    unsigned short* cwp    = (unsigned short*)alloc((size_t)LAYERS * H * H * 2);
    float*          z1     = (float*)         alloc((size_t)B * H * 4);
    float*          z2     = (float*)         alloc((size_t)B * (H / 2) * 4);

    hipMemsetAsync(pooled, 0, (size_t)LAYERS * B * H * 4, stream);
    hipMemsetAsync(counts, 0, (size_t)B * 4, stream);

    count_kernel<<<(N + 255) / 256, 256, 0, stream>>>(batch, counts, N);
    pack_cw_kernel<<<(LAYERS * H * H + 255) / 256, 256, 0, stream>>>(cw, cwp);
    embed_kernel<<<N, H, 0, stream>>>(x, emb_w, emb_b, emb_lng, emb_lnb, h, hbm);

    const int etiles = (int)((E + TE - 1) / TE);
    int grid = etiles < 2048 ? etiles : 2048;
    if (grid < 1) grid = 1;

    for (int l = 0; l < LAYERS; ++l) {
        hipMemsetAsync(abuf, 0, (size_t)N * H * 4, stream);
        msg_kernel<<<grid, 256, 0, stream>>>(hbm, ei, ea, cw, cb, cwp,
                                             abuf, E, etiles, l);
        ln_res_kernel<<<N, H, 0, stream>>>(h, hbm, abuf, ng, nb, batch, pooled, l, B);
    }

    head1_kernel<<<B, H, 0, stream>>>(pooled, counts, hw1, hb1, z1, B);
    head2_kernel<<<B, H / 2, 0, stream>>>(z1, hw2, hb2, z2);
    head3_kernel<<<B, H / 2, 0, stream>>>(z2, hw3, hb3, out);
}
